// PoseFeatureExtractor_23983097381097
// MI455X (gfx1250) — matrix operand order repaired and checked by reading, not compiled, not run on hardware
//
#include <hip/hip_runtime.h>
#include <math.h>

// ---------------- problem constants ----------------
#define BB   64
#define SS   2048
#define JJ   25
#define ROWS (BB * SS)          // 131072
#define IN_MLP 300
#define KPAD   320              // 300 padded to multiple of 32
#define HID    256
#define FEAT   128
#define MTILE  32               // rows per workgroup

typedef __attribute__((ext_vector_type(16))) _Float16 v16h;
typedef __attribute__((ext_vector_type(8)))  float    v8f;
typedef __attribute__((ext_vector_type(4)))  _Float16 v4h;

union H16 { v16h v; v4h q[4]; };

__constant__ int d_c0[13] = {0,1,1,5,7,6,8,1,1,11,13,12,14};
__constant__ int d_c1[13] = {1,5,6,7,9,8,10,11,12,13,15,14,16};

// ---------------------------------------------------------------------------
// Prep kernel: W1 -> f16 transposed [HID][KPAD], W2 -> f16 transposed
// [FEAT][HID], BN folded with b1 into scale/shift.
// ---------------------------------------------------------------------------
__global__ void prep_kernel(const float* __restrict__ W1, const float* __restrict__ b1,
                            const float* __restrict__ gamma, const float* __restrict__ beta,
                            const float* __restrict__ rmean, const float* __restrict__ rvar,
                            const float* __restrict__ W2,
                            _Float16* __restrict__ W1T, _Float16* __restrict__ W2T,
                            float* __restrict__ scale, float* __restrict__ shift) {
    int idx = blockIdx.x * blockDim.x + threadIdx.x;
    const int N1 = HID * KPAD;          // 81920
    const int N2 = FEAT * HID;          // 32768
    if (idx < N1) {
        int n = idx / KPAD, k = idx % KPAD;
        W1T[idx] = (k < IN_MLP) ? (_Float16)W1[k * HID + n] : (_Float16)0.0f;
    } else if (idx < N1 + N2) {
        int i = idx - N1;
        int n = i / HID, k = i % HID;
        W2T[i] = (_Float16)W2[k * FEAT + n];
    } else if (idx < N1 + N2 + HID) {
        int n = idx - N1 - N2;
        float s = gamma[n] * rsqrtf(rvar[n] + 1e-5f);
        scale[n] = s;
        shift[n] = beta[n] + s * (b1[n] - rmean[n]);
    }
}

// ---------------------------------------------------------------------------
// Fused: features -> LDS(f16) -> WMMA GEMM1 -> BN+ReLU -> LDS -> WMMA GEMM2
// One workgroup (8 wave32) handles 32 consecutive (b,s) rows.
// ---------------------------------------------------------------------------
__global__ void __launch_bounds__(256)
pose_mlp_kernel(const float* __restrict__ poses, const float* __restrict__ jwg,
                const _Float16* __restrict__ W1T, const _Float16* __restrict__ W2T,
                const float* __restrict__ scale, const float* __restrict__ shift,
                const float* __restrict__ b2, float* __restrict__ out) {
    __shared__ _Float16 sX [MTILE][KPAD];   // 20480 B  feature matrix (f16)
    __shared__ _Float16 sX2[MTILE][HID];    // 16384 B  hidden activations (f16)
    __shared__ float    sPn[MTILE][76];     //  9728 B  normalized pose @ frame t
    __shared__ float    sRoot[3][MTILE][4]; //  1536 B
    __shared__ float    sInv [3][MTILE];    //   384 B

    const int tid     = threadIdx.x;
    const int wave    = tid >> 5;
    const int lane    = tid & 31;
    const int lane15  = lane & 15;
    const int halfsel = lane >> 4;          // 0 or 1
    const int row_base = blockIdx.x * MTILE;

    // ---- phase 1: per (row, frame) root + 1/(max_dist+1e-8) ----
    for (int task = tid; task < MTILE * 3; task += 256) {
        int r = task / 3, f = task % 3;
        int row = row_base + r;
        int t = row & (SS - 1);
        int tf = t - f; if (tf < 0) tf = 0;
        const float* fp = poses + ((size_t)(row - t + tf) * JJ) * 3;
        float rx = fp[0], ry = fp[1], rz = fp[2];
        float maxd = 0.0f;
        for (int j = 0; j < JJ; ++j) {
            float dx = fp[j*3+0] - rx, dy = fp[j*3+1] - ry, dz = fp[j*3+2] - rz;
            maxd = fmaxf(maxd, sqrtf(dx*dx + dy*dy + dz*dz));
        }
        sRoot[f][r][0] = rx; sRoot[f][r][1] = ry; sRoot[f][r][2] = rz;
        sInv[f][r] = 1.0f / (maxd + 1e-8f);
    }
    __syncthreads();

    // ---- phase 2: per (row, joint) pos/vel/acc features + pn(t) ----
    for (int task = tid; task < MTILE * JJ; task += 256) {
        int r = task / JJ, j = task % JJ;
        int row = row_base + r;
        int t = row & (SS - 1);
        float jw = jwg[j];
        float pf[3][3];
        #pragma unroll
        for (int f = 0; f < 3; ++f) {
            int tf = t - f; if (tf < 0) tf = 0;
            const float* fp = poses + ((size_t)(row - t + tf) * JJ + j) * 3;
            float inv = sInv[f][r];
            pf[f][0] = (fp[0] - sRoot[f][r][0]) * inv;
            pf[f][1] = (fp[1] - sRoot[f][r][1]) * inv;
            pf[f][2] = (fp[2] - sRoot[f][r][2]) * inv;
        }
        bool has1 = (t >= 1), has2 = (t >= 2);
        #pragma unroll
        for (int c = 0; c < 3; ++c) {
            float p0  = pf[0][c];
            float v0  = has1 ? (p0 - pf[1][c]) : 0.0f;
            float vm1 = has2 ? (pf[1][c] - pf[2][c]) : 0.0f;
            float a0  = has1 ? (v0 - vm1) : 0.0f;
            sPn[r][j*3 + c] = p0;
            sX[r][j*12 + 0 + c] = (_Float16)(p0 * jw);
            sX[r][j*12 + 3 + c] = (_Float16)(v0 * jw);
            sX[r][j*12 + 6 + c] = (_Float16)(a0 * jw);
        }
    }
    // zero the K padding columns 300..319
    for (int i = tid; i < MTILE * (KPAD - IN_MLP); i += 256)
        sX[i / (KPAD - IN_MLP)][IN_MLP + i % (KPAD - IN_MLP)] = (_Float16)0.0f;
    __syncthreads();

    // ---- phase 3: joint angles (needs all pn(t) of the row) ----
    for (int task = tid; task < MTILE * JJ; task += 256) {
        int r = task / JJ, j = task % JJ;
        if (j < 13) {
            float jw = jwg[j];
            int a0 = d_c0[j], a1 = d_c1[j];
            float vx = sPn[r][a1*3+0] - sPn[r][a0*3+0];
            float vy = sPn[r][a1*3+1] - sPn[r][a0*3+1];
            float vz = sPn[r][a1*3+2] - sPn[r][a0*3+2];
            float nrm = fmaxf(sqrtf(vx*vx + vy*vy + vz*vz), 1e-12f);
            float invn = 1.0f / nrm;
            float v3[3] = {vx, vy, vz};
            #pragma unroll
            for (int c = 0; c < 3; ++c) {
                float val = fminf(fmaxf(v3[c] * invn, -1.0f), 1.0f);
                sX[r][j*12 + 9 + c] = (_Float16)(acosf(val) * jw);
            }
        } else {
            sX[r][j*12 +  9] = (_Float16)0.0f;
            sX[r][j*12 + 10] = (_Float16)0.0f;
            sX[r][j*12 + 11] = (_Float16)0.0f;
        }
    }
    __syncthreads();

    // ---- phase 4: GEMM1  C1[32x256] = X[32x320(f16)] * W1T^T, WMMA f16 ----
    // wave w: mt = w&1 (2 M-tiles), covers 4 N-tiles starting at (w>>1)*4
    const int mt = wave & 1;
    const int ng = wave >> 1;
    v8f acc1[4];
    {
        v8f z = {};
        acc1[0] = z; acc1[1] = z; acc1[2] = z; acc1[3] = z;
    }
    const int mrow = mt * 16 + lane15;
    for (int ks = 0; ks < KPAD / 32; ++ks) {
        int k0 = ks * 32;
        H16 afr;
        #pragma unroll
        for (int g = 0; g < 4; ++g) {   // A: k = k0 + (g/2)*16 + (g%2)*4 + half*8
            int k = k0 + (g >> 1) * 16 + (g & 1) * 4 + halfsel * 8;
            afr.q[g] = *(const v4h*)&sX[mrow][k];
        }
        #pragma unroll
        for (int i = 0; i < 4; ++i) {
            int n = (ng * 4 + i) * 16 + lane15;
            const _Float16* wrow = W1T + (size_t)n * KPAD;
            H16 bfr;
            #pragma unroll
            for (int g = 0; g < 4; ++g) {  // B: k = k0 + (g/2)*16 + (g%2)*4 + half*8
                int k = k0 + (g >> 1) * 16 + (g & 1) * 4 + halfsel * 8;
                bfr.q[g] = *(const v4h*)&wrow[k];
            }
            acc1[i] = __builtin_amdgcn_wmma_f32_16x16x32_f16(
                false, afr.v, false, bfr.v, (short)0, acc1[i], false, false);
        }
    }
    // BN (scale/shift, b1 folded) + ReLU -> f16 LDS
    #pragma unroll
    for (int i = 0; i < 4; ++i) {
        int n = (ng * 4 + i) * 16 + lane15;
        float sc = scale[n], sh = shift[n];
        #pragma unroll
        for (int v = 0; v < 8; ++v) {
            int m = mt * 16 + v + halfsel * 8;
            float val = fmaxf(acc1[i][v] * sc + sh, 0.0f);
            sX2[m][n] = (_Float16)val;
        }
    }
    __syncthreads();

    // ---- phase 5: GEMM2  C2[32x128] = X2[32x256] * W2T^T ----
    const int mt2 = wave & 1;
    const int ng2 = wave >> 1;   // each covers 2 N-tiles
    v8f acc2[2];
    {
        v8f z = {};
        acc2[0] = z; acc2[1] = z;
    }
    const int mrow2 = mt2 * 16 + lane15;
    for (int ks = 0; ks < HID / 32; ++ks) {
        int k0 = ks * 32;
        H16 afr;
        #pragma unroll
        for (int g = 0; g < 4; ++g) {
            int k = k0 + (g >> 1) * 16 + (g & 1) * 4 + halfsel * 8;
            afr.q[g] = *(const v4h*)&sX2[mrow2][k];
        }
        #pragma unroll
        for (int i = 0; i < 2; ++i) {
            int n = (ng2 * 2 + i) * 16 + lane15;
            const _Float16* wrow = W2T + (size_t)n * HID;
            H16 bfr;
            #pragma unroll
            for (int g = 0; g < 4; ++g) {
                int k = k0 + (g >> 1) * 16 + (g & 1) * 4 + halfsel * 8;
                bfr.q[g] = *(const v4h*)&wrow[k];
            }
            acc2[i] = __builtin_amdgcn_wmma_f32_16x16x32_f16(
                false, afr.v, false, bfr.v, (short)0, acc2[i], false, false);
        }
    }
    // epilogue: + b2, store fp32
    #pragma unroll
    for (int i = 0; i < 2; ++i) {
        int n = (ng2 * 2 + i) * 16 + lane15;
        float bb = b2[n];
        #pragma unroll
        for (int v = 0; v < 8; ++v) {
            int m = mt2 * 16 + v + halfsel * 8;
            size_t row = (size_t)(row_base + m);
            out[row * FEAT + n] = acc2[i][v] + bb;
        }
    }
}

// ---------------------------------------------------------------------------
extern "C" void kernel_launch(void* const* d_in, const int* in_sizes, int n_in,
                              void* d_out, int out_size, void* d_ws, size_t ws_size,
                              hipStream_t stream) {
    const float* poses = (const float*)d_in[0];
    const float* W1    = (const float*)d_in[1];
    const float* b1    = (const float*)d_in[2];
    const float* gamma = (const float*)d_in[3];
    const float* beta  = (const float*)d_in[4];
    const float* rmean = (const float*)d_in[5];
    const float* rvar  = (const float*)d_in[6];
    const float* W2    = (const float*)d_in[7];
    const float* b2    = (const float*)d_in[8];
    const float* jw    = (const float*)d_in[9];

    // workspace layout (needs ~226 KB):
    _Float16* W1T   = (_Float16*)d_ws;            // [256][320] f16
    _Float16* W2T   = W1T + (size_t)HID * KPAD;   // [128][256] f16
    float*    scale = (float*)(W2T + (size_t)FEAT * HID);
    float*    shift = scale + HID;

    const int prep_elems = HID * KPAD + FEAT * HID + HID;   // 114944
    prep_kernel<<<(prep_elems + 255) / 256, 256, 0, stream>>>(
        W1, b1, gamma, beta, rmean, rvar, W2, W1T, W2T, scale, shift);

    pose_mlp_kernel<<<ROWS / MTILE, 256, 0, stream>>>(
        poses, jw, W1T, W2T, scale, shift, b2, (float*)d_out);
}
